// XFeatWrapper_4810363372650
// MI455X (gfx1250) — matrix operand order repaired and checked by reading, not compiled, not run on hardware
//
#include <hip/hip_runtime.h>
#include <math.h>

typedef __attribute__((ext_vector_type(16))) _Float16 v16h;
typedef __attribute__((ext_vector_type(8)))  float    v8f;

// =====================================================================
// helpers
// =====================================================================
__device__ __forceinline__ unsigned long long score_key(float v, int i) {
  unsigned u = __float_as_uint(v);
  unsigned k = (u & 0x80000000u) ? ~u : (u | 0x80000000u);
  return (((unsigned long long)k) << 32) | (unsigned)(~(unsigned)i);
}

// =====================================================================
// instance norm (deterministic 2-stage reduction)
// =====================================================================
__global__ void k_stats_partial(const float* __restrict__ x, int n, float* __restrict__ part) {
  __shared__ float s0[256], s1[256];
  float a = 0.f, b = 0.f;
  for (int i = blockIdx.x * blockDim.x + threadIdx.x; i < n; i += gridDim.x * blockDim.x) {
    float v = x[i]; a += v; b += v * v;
  }
  s0[threadIdx.x] = a; s1[threadIdx.x] = b;
  __syncthreads();
  for (int s = 128; s > 0; s >>= 1) {
    if ((int)threadIdx.x < s) { s0[threadIdx.x] += s0[threadIdx.x + s]; s1[threadIdx.x] += s1[threadIdx.x + s]; }
    __syncthreads();
  }
  if (threadIdx.x == 0) { part[blockIdx.x] = s0[0]; part[1024 + blockIdx.x] = s1[0]; }
}

__global__ void k_stats_final(const float* __restrict__ part, float* __restrict__ stats, int n) {
  __shared__ float s0[256], s1[256];
  float a = 0.f, b = 0.f;
  for (int i = threadIdx.x; i < 1024; i += 256) { a += part[i]; b += part[1024 + i]; }
  s0[threadIdx.x] = a; s1[threadIdx.x] = b;
  __syncthreads();
  for (int s = 128; s > 0; s >>= 1) {
    if ((int)threadIdx.x < s) { s0[threadIdx.x] += s0[threadIdx.x + s]; s1[threadIdx.x] += s1[threadIdx.x + s]; }
    __syncthreads();
  }
  if (threadIdx.x == 0) {
    float mu  = s0[0] / (float)n;
    float var = s1[0] / (float)n - mu * mu;
    stats[0] = mu; stats[1] = rsqrtf(var + 1e-5f);
  }
}

__global__ void k_instnorm(const float* __restrict__ x, const float* __restrict__ stats,
                           float* __restrict__ y, int n) {
  int i = blockIdx.x * blockDim.x + threadIdx.x;
  if (i < n) y[i] = (x[i] - stats[0]) * stats[1];
}

// =====================================================================
// BN folding:  scale = g*rsqrt(v+eps), shift = b - m*scale
// =====================================================================
__global__ void k_bn_prep(const float* g, const float* b, const float* m, const float* v,
                          float* sc, float* sh, int c) {
  int i = blockIdx.x * blockDim.x + threadIdx.x;
  if (i < c) { float s = g[i] * rsqrtf(v[i] + 1e-5f); sc[i] = s; sh[i] = b[i] - m[i] * s; }
}
__global__ void k_bias_prep(const float* b, float* sc, float* sh, int c) {
  int i = blockIdx.x * blockDim.x + threadIdx.x;
  if (i < c) { sc[i] = 1.f; sh[i] = b[i]; }
}

// =====================================================================
// Weight repack: OIHW f32 -> per-(cout-tile, k-chunk) A-fragments in the
// CDNA5 16-bit A 16x32 per-lane layout (16 halves per lane, 32 lanes).
// =====================================================================
__global__ void k_repack(const float* __restrict__ w, _Float16* __restrict__ wp,
                         int Cout, int Ktot) {
  int nkc = (Ktot + 31) >> 5;
  int nct = (Cout + 15) >> 4;
  int total = nct * nkc * 32;
  int id = blockIdx.x * blockDim.x + threadIdx.x;
  if (id >= total) return;
  int lane = id & 31;
  int tile = id >> 5;
  int ct = tile / nkc, kc = tile - ct * nkc;
  int m = ct * 16 + (lane & 15);
  int aKoff = (lane < 16) ? 0 : 8;
  for (int j = 0; j < 8; ++j) {
    int k0 = kc * 32 + 16 * (j >> 2) + aKoff + (j & 3) * 2;
    float a0 = (m < Cout && k0     < Ktot) ? w[(size_t)m * Ktot + k0]     : 0.f;
    float a1 = (m < Cout && k0 + 1 < Ktot) ? w[(size_t)m * Ktot + k0 + 1] : 0.f;
    wp[(size_t)id * 16 + 2 * j]     = (_Float16)a0;
    wp[(size_t)id * 16 + 2 * j + 1] = (_Float16)a1;
  }
}

// =====================================================================
// General implicit-GEMM conv (any k/stride/pad) via V_WMMA_F32_16X16X32_F16.
// One wave owns a 16(Cout) x 32(pixels) tile: 2 accumulators share 1 A frag.
// =====================================================================
__device__ __forceinline__ float in_elem(const float* __restrict__ x,
                                         int Hin, int Win, int ks, int kk2, int Ktot,
                                         int iy0, int ix0, int kk) {
  if (kk >= Ktot) return 0.f;
  int ci  = kk / kk2;
  int rem = kk - ci * kk2;
  int r = rem / ks, s = rem - r * ks;
  int iy = iy0 + r, ix = ix0 + s;
  if ((unsigned)iy >= (unsigned)Hin || (unsigned)ix >= (unsigned)Win) return 0.f;
  return x[((size_t)ci * Hin + iy) * Win + ix];
}

__global__ __launch_bounds__(32)
void k_conv_wmma(const float* __restrict__ x, const _Float16* __restrict__ wp,
                 const float* __restrict__ scale, const float* __restrict__ shift,
                 float* __restrict__ y,
                 int Cin, int Cout, int Hin, int Win, int Hout, int Wout,
                 int ks, int stride, int pad, int relu) {
  const int lane     = threadIdx.x;
  const int pixBase  = blockIdx.x * 32;
  const int ct       = blockIdx.y;
  const int coutBase = ct * 16;
  const int npix = Hout * Wout;
  const int kk2  = ks * ks;
  const int Ktot = Cin * kk2;
  const int nkc  = (Ktot + 31) >> 5;

  const int nCol = lane & 15;
  int p0 = pixBase + nCol, p1 = p0 + 16;
  int vp0 = (p0 < npix), vp1 = (p1 < npix);
  int q0 = vp0 ? p0 : 0, q1 = vp1 ? p1 : 0;
  int oy0 = q0 / Wout, ox0 = q0 - oy0 * Wout;
  int oy1 = q1 / Wout, ox1 = q1 - oy1 * Wout;
  int iyA = oy0 * stride - pad, ixA = ox0 * stride - pad;
  int iyB = oy1 * stride - pad, ixB = ox1 * stride - pad;
  const int bKoff = (lane < 16) ? 0 : 8;

  v8f acc0 = {}, acc1 = {};
  for (int kcIdx = 0; kcIdx < nkc; ++kcIdx) {
    int kc = kcIdx << 5;
    const uint4* ap = (const uint4*)(wp + ((size_t)(ct * nkc + kcIdx) * 32 + lane) * 16);
    __builtin_prefetch(ap + 64, 0, 0);
    union { uint4 u[2]; v16h h; } au;
    au.u[0] = ap[0]; au.u[1] = ap[1];
    v16h afrag = au.h;

    v16h b0 = {}, b1 = {};
#pragma unroll
    for (int j = 0; j < 8; ++j) {
      int k0 = kc + 16 * (j >> 2) + bKoff + (j & 3) * 2;
      float e0 = 0.f, e1 = 0.f, f0 = 0.f, f1 = 0.f;
      if (vp0) {
        e0 = in_elem(x, Hin, Win, ks, kk2, Ktot, iyA, ixA, k0);
        e1 = in_elem(x, Hin, Win, ks, kk2, Ktot, iyA, ixA, k0 + 1);
      }
      if (vp1) {
        f0 = in_elem(x, Hin, Win, ks, kk2, Ktot, iyB, ixB, k0);
        f1 = in_elem(x, Hin, Win, ks, kk2, Ktot, iyB, ixB, k0 + 1);
      }
      b0[2 * j] = (_Float16)e0; b0[2 * j + 1] = (_Float16)e1;
      b1[2 * j] = (_Float16)f0; b1[2 * j + 1] = (_Float16)f1;
    }
    acc0 = __builtin_amdgcn_wmma_f32_16x16x32_f16(false, afrag, false, b0,
                                                  (short)0, acc0, false, false);
    acc1 = __builtin_amdgcn_wmma_f32_16x16x32_f16(false, afrag, false, b1,
                                                  (short)0, acc1, false, false);
  }

#pragma unroll
  for (int v = 0; v < 8; ++v) {
    int m = coutBase + v + ((lane < 16) ? 0 : 8);
    if (m < Cout) {
      float s = scale[m], h = shift[m];
      float o0 = acc0[v] * s + h;
      float o1 = acc1[v] * s + h;
      if (relu) { o0 = fmaxf(o0, 0.f); o1 = fmaxf(o1, 0.f); }
      if (vp0) y[(size_t)m * npix + p0] = o0;
      if (vp1) y[(size_t)m * npix + p1] = o1;
    }
  }
}

// =====================================================================
// 1x1 stride-1 conv (pure GEMM): async global->LDS staging of the B tile
// (32 K x 64 pixels, f32) via global_load_async_to_lds_b128, then 4 WMMAs
// per chunk sharing one A fragment.
// =====================================================================
__global__ __launch_bounds__(32)
void k_conv1x1_async(const float* __restrict__ x, const _Float16* __restrict__ wp,
                     const float* __restrict__ scale, const float* __restrict__ shift,
                     float* __restrict__ y, int Cin, int Cout, int npix, int relu) {
  __shared__ float sB[32 * 64];      // [k][pixel]
  const int lane     = threadIdx.x;
  const int pixBase  = blockIdx.x * 64;
  const int ct       = blockIdx.y;
  const int coutBase = ct * 16;
  const int nkc      = (Cin + 31) >> 5;
  const int bKoff    = (lane < 16) ? 0 : 8;
  const int nCol     = lane & 15;

  v8f acc0 = {}, acc1 = {}, acc2 = {}, acc3 = {};
  for (int kcIdx = 0; kcIdx < nkc; ++kcIdx) {
    int c0 = kcIdx << 5;
    // previous chunk's LDS reads must drain before we overwrite the tile
    asm volatile("s_wait_dscnt 0x0" ::: "memory");
    // 512 x 16B segments, 32 lanes -> 16 async issues
#pragma unroll
    for (int r = 0; r < 16; ++r) {
      int seg  = r * 32 + lane;
      int k    = seg >> 4;          // row (K)
      int part = seg & 15;          // 16B segment within 64-pixel row
      const float* gp = x + (size_t)(c0 + k) * npix + pixBase + part * 4;
      unsigned ldsa = (unsigned)(size_t)(&sB[k * 64 + part * 4]);
      asm volatile("global_load_async_to_lds_b128 %0, %1, off"
                   :: "v"(ldsa), "v"((unsigned long long)(size_t)gp)
                   : "memory");
    }
    asm volatile("s_wait_asynccnt 0x0" ::: "memory");

    const uint4* ap = (const uint4*)(wp + ((size_t)(ct * nkc + kcIdx) * 32 + lane) * 16);
    union { uint4 u[2]; v16h h; } au;
    au.u[0] = ap[0]; au.u[1] = ap[1];
    v16h afrag = au.h;

    v16h b0 = {}, b1 = {}, b2 = {}, b3 = {};
#pragma unroll
    for (int j = 0; j < 8; ++j) {
      int k0 = 16 * (j >> 2) + bKoff + (j & 3) * 2;
      b0[2 * j]     = (_Float16)sB[k0 * 64 + nCol];
      b0[2 * j + 1] = (_Float16)sB[(k0 + 1) * 64 + nCol];
      b1[2 * j]     = (_Float16)sB[k0 * 64 + 16 + nCol];
      b1[2 * j + 1] = (_Float16)sB[(k0 + 1) * 64 + 16 + nCol];
      b2[2 * j]     = (_Float16)sB[k0 * 64 + 32 + nCol];
      b2[2 * j + 1] = (_Float16)sB[(k0 + 1) * 64 + 32 + nCol];
      b3[2 * j]     = (_Float16)sB[k0 * 64 + 48 + nCol];
      b3[2 * j + 1] = (_Float16)sB[(k0 + 1) * 64 + 48 + nCol];
    }
    acc0 = __builtin_amdgcn_wmma_f32_16x16x32_f16(false, afrag, false, b0, (short)0, acc0, false, false);
    acc1 = __builtin_amdgcn_wmma_f32_16x16x32_f16(false, afrag, false, b1, (short)0, acc1, false, false);
    acc2 = __builtin_amdgcn_wmma_f32_16x16x32_f16(false, afrag, false, b2, (short)0, acc2, false, false);
    acc3 = __builtin_amdgcn_wmma_f32_16x16x32_f16(false, afrag, false, b3, (short)0, acc3, false, false);
  }

#pragma unroll
  for (int v = 0; v < 8; ++v) {
    int m = coutBase + v + ((lane < 16) ? 0 : 8);
    if (m < Cout) {
      float s = scale[m], h = shift[m];
      int pc = pixBase + nCol;
      float o0 = acc0[v] * s + h, o1 = acc1[v] * s + h;
      float o2 = acc2[v] * s + h, o3 = acc3[v] * s + h;
      if (relu) { o0 = fmaxf(o0, 0.f); o1 = fmaxf(o1, 0.f); o2 = fmaxf(o2, 0.f); o3 = fmaxf(o3, 0.f); }
      if (pc      < npix) y[(size_t)m * npix + pc]      = o0;
      if (pc + 16 < npix) y[(size_t)m * npix + pc + 16] = o1;
      if (pc + 32 < npix) y[(size_t)m * npix + pc + 32] = o2;
      if (pc + 48 < npix) y[(size_t)m * npix + pc + 48] = o3;
    }
  }
}

// =====================================================================
// misc elementwise / structural kernels
// =====================================================================
__global__ void k_add(const float* a, const float* b, float* y, int n) {
  int i = blockIdx.x * blockDim.x + threadIdx.x;
  if (i < n) y[i] = a[i] + b[i];
}
__global__ void k_add3(const float* a, const float* b, const float* c, float* y, int n) {
  int i = blockIdx.x * blockDim.x + threadIdx.x;
  if (i < n) y[i] = a[i] + b[i] + c[i];
}
__global__ void k_sigmoid(float* x, int n) {
  int i = blockIdx.x * blockDim.x + threadIdx.x;
  if (i < n) x[i] = 1.f / (1.f + expf(-x[i]));
}

__global__ void k_avgpool4(const float* __restrict__ x, float* __restrict__ y, int Hin, int Win) {
  int Ho = Hin >> 2, Wo = Win >> 2;
  int i = blockIdx.x * blockDim.x + threadIdx.x;
  if (i >= Ho * Wo) return;
  int oy = i / Wo, ox = i - oy * Wo;
  float s = 0.f;
  for (int dy = 0; dy < 4; ++dy)
    for (int dx = 0; dx < 4; ++dx)
      s += x[(oy * 4 + dy) * Win + ox * 4 + dx];
  y[i] = s * (1.f / 16.f);
}

__global__ void k_unfold8(const float* __restrict__ x, float* __restrict__ y) {
  int i = blockIdx.x * blockDim.x + threadIdx.x;      // 64*19200
  if (i >= 64 * 19200) return;
  int c = i / 19200;
  int p = i - c * 19200;
  int by = p / 160, bx = p - by * 160;
  int wy = c >> 3, wx = c & 7;
  y[i] = x[(by * 8 + wy) * 1280 + bx * 8 + wx];
}

__global__ void k_resize(const float* __restrict__ x, float* __restrict__ y,
                         int C, int Hin, int Win, int Hout, int Wout) {
  int i = blockIdx.x * blockDim.x + threadIdx.x;
  int hw = Hout * Wout;
  if (i >= C * hw) return;
  int c = i / hw;
  int r = i - c * hw;
  int oy = r / Wout, ox = r - oy * Wout;
  float sy = (oy + 0.5f) * (float)Hin / (float)Hout - 0.5f;
  float sx = (ox + 0.5f) * (float)Win / (float)Wout - 0.5f;
  float fy = floorf(sy), fx = floorf(sx);
  float ty = sy - fy, tx = sx - fx;
  int y0 = (int)fy, x0 = (int)fx, y1 = y0 + 1, x1 = x0 + 1;
  y0 = min(max(y0, 0), Hin - 1); y1 = min(max(y1, 0), Hin - 1);
  x0 = min(max(x0, 0), Win - 1); x1 = min(max(x1, 0), Win - 1);
  const float* f = x + (size_t)c * Hin * Win;
  y[i] = (1.f - ty) * ((1.f - tx) * f[y0 * Win + x0] + tx * f[y0 * Win + x1])
       +         ty * ((1.f - tx) * f[y1 * Win + x0] + tx * f[y1 * Win + x1]);
}

// softmax over 65 channels + 8x8 pixel-shuffle into full-res heatmap
__global__ void k_kpts_heat(const float* __restrict__ klog, float* __restrict__ k1h) {
  int p = blockIdx.x * blockDim.x + threadIdx.x;
  if (p >= 19200) return;
  int by = p / 160, bx = p - by * 160;
  float mx = -1e30f;
  for (int c = 0; c < 65; ++c) mx = fmaxf(mx, klog[c * 19200 + p]);
  float sum = 0.f;
  for (int c = 0; c < 65; ++c) sum += expf(klog[c * 19200 + p] - mx);
  float inv = 1.f / sum;
  for (int c = 0; c < 64; ++c) {
    float v = expf(klog[c * 19200 + p] - mx) * inv;
    int wy = c >> 3, wx = c & 7;
    k1h[(by * 8 + wy) * 1280 + bx * 8 + wx] = v;
  }
}

__global__ void k_maxpool5(const float* __restrict__ x, float* __restrict__ y, int H, int W) {
  int p = blockIdx.x * blockDim.x + threadIdx.x;
  if (p >= H * W) return;
  int py = p / W, px = p - py * W;
  float m = -INFINITY;
  for (int dy = -2; dy <= 2; ++dy) {
    int yy = py + dy; if ((unsigned)yy >= (unsigned)H) continue;
    for (int dx = -2; dx <= 2; ++dx) {
      int xx = px + dx; if ((unsigned)xx >= (unsigned)W) continue;
      m = fmaxf(m, x[yy * W + xx]);
    }
  }
  y[p] = m;
}

// bilinear sample of 120x160 heatmap at every full-res pixel (grid_sample semantics)
__global__ void k_rel(const float* __restrict__ hmap, float* __restrict__ rel) {
  int p = blockIdx.x * blockDim.x + threadIdx.x;
  if (p >= 1228800) return;
  int py = p / 1280, px = p - py * 1280;
  float ix = (float)px * (160.f / 1279.f) - 0.5f;
  float iy = (float)py * (120.f / 959.f) - 0.5f;
  float fx = floorf(ix), fy = floorf(iy);
  float tx = ix - fx, ty = iy - fy;
  int x0 = (int)fx, y0 = (int)fy;
  float out = 0.f;
  for (int dy = 0; dy < 2; ++dy) {
    float wy = dy ? ty : (1.f - ty);
    int yy = y0 + dy; if ((unsigned)yy >= 120u) continue;
    for (int dx = 0; dx < 2; ++dx) {
      float wx = dx ? tx : (1.f - tx);
      int xx = x0 + dx; if ((unsigned)xx >= 160u) continue;
      out += wy * wx * hmap[yy * 160 + xx];
    }
  }
  rel[p] = out;
}

__global__ void k_scores(const float* k1h, const float* lm, const float* rel, float* sc, int n) {
  int i = blockIdx.x * blockDim.x + threadIdx.x;
  if (i >= n) return;
  bool mask = (k1h[i] == lm[i]) && (k1h[i] > 0.05f);
  sc[i] = mask ? k1h[i] * rel[i] : -1.0f;
}

__global__ void k_normfeat(const float* __restrict__ feats, float* __restrict__ m1) {
  int p = blockIdx.x * blockDim.x + threadIdx.x;
  if (p >= 19200) return;
  float s = 0.f;
  for (int c = 0; c < 64; ++c) { float v = feats[c * 19200 + p]; s += v * v; }
  float inv = 1.f / fmaxf(sqrtf(s), 1e-12f);
  for (int c = 0; c < 64; ++c) m1[c * 19200 + p] = feats[c * 19200 + p] * inv;
}

// =====================================================================
// deterministic top-2000: radix-select on unique 64-bit keys
// =====================================================================
__global__ void k_sel_init(unsigned long long* sel) { sel[0] = 0ull; sel[1] = 2000ull; }

__global__ void k_hist64(const float* __restrict__ s, int n,
                         const unsigned long long* __restrict__ sel, int level,
                         unsigned* __restrict__ hist) {
  unsigned long long prefix = sel[0];
  for (int i = blockIdx.x * blockDim.x + threadIdx.x; i < n; i += gridDim.x * blockDim.x) {
    unsigned long long key = score_key(s[i], i);
    bool ok = (level == 0) || ((key >> (64 - 8 * level)) == prefix);
    if (ok) atomicAdd(&hist[(unsigned)((key >> (56 - 8 * level)) & 255ull)], 1u);
  }
}

__global__ void k_pick64(const unsigned* __restrict__ hist, unsigned long long* __restrict__ sel) {
  if (threadIdx.x == 0 && blockIdx.x == 0) {
    unsigned long long prefix = sel[0];
    unsigned need = (unsigned)sel[1];
    unsigned cum = 0; int chosen = 0;
    for (int b = 255; b >= 0; --b) {
      unsigned h = hist[b];
      if (cum + h >= need) { chosen = b; break; }
      cum += h;
    }
    sel[0] = (prefix << 8) | (unsigned long long)(unsigned)chosen;
    sel[1] = (unsigned long long)(need - cum);
  }
}

__global__ void k_compact64(const float* __restrict__ s, int n,
                            const unsigned long long* __restrict__ sel,
                            unsigned long long* __restrict__ cand, unsigned* __restrict__ cnt) {
  unsigned long long T = sel[0];
  for (int i = blockIdx.x * blockDim.x + threadIdx.x; i < n; i += gridDim.x * blockDim.x) {
    unsigned long long key = score_key(s[i], i);
    if (key >= T) {
      unsigned p = atomicAdd(cnt, 1u);
      if (p < 2048u) cand[p] = key;
    }
  }
}

__global__ __launch_bounds__(1024) void k_bitonic2048(unsigned long long* cand) {
  __shared__ unsigned long long s[2048];
  int t = threadIdx.x;
  s[t] = cand[t]; s[t + 1024] = cand[t + 1024];
  __syncthreads();
  for (int k = 2; k <= 2048; k <<= 1) {
    for (int j = k >> 1; j > 0; j >>= 1) {
      for (int i = t; i < 2048; i += 1024) {
        int ixj = i ^ j;
        if (ixj > i) {
          bool desc = ((i & k) == 0);
          unsigned long long a = s[i], b = s[ixj];
          if ((a < b) == desc) { s[i] = b; s[ixj] = a; }
        }
      }
      __syncthreads();
    }
  }
  cand[t] = s[t]; cand[t + 1024] = s[t + 1024];
}

__global__ void k_emit(const unsigned long long* __restrict__ cand,
                       float* __restrict__ mk, float* __restrict__ sc) {
  int i = blockIdx.x * blockDim.x + threadIdx.x;
  if (i >= 2000) return;
  unsigned long long key = cand[i];
  unsigned idx = ~(unsigned)(key & 0xffffffffull);
  unsigned k32 = (unsigned)(key >> 32);
  unsigned u = (k32 & 0x80000000u) ? (k32 & 0x7fffffffu) : ~k32;
  sc[i] = __uint_as_float(u);
  mk[2 * i]     = (float)(idx % 1280u);
  mk[2 * i + 1] = (float)(idx / 1280u);
}

// =====================================================================
// bicubic descriptor sampling + L2 normalization (block per keypoint)
// =====================================================================
__device__ __forceinline__ float cubw(float t) {
  t = fabsf(t);
  const float a = -0.75f;
  if (t <= 1.f) return ((a + 2.f) * t - (a + 3.f)) * t * t + 1.f;
  if (t < 2.f)  return a * (((t - 5.f) * t + 8.f) * t - 4.f);
  return 0.f;
}

__global__ __launch_bounds__(64) void k_desc(const float* __restrict__ m1,
                                             const float* __restrict__ kpts,
                                             float* __restrict__ descs) {
  int kp = blockIdx.x;
  int c  = threadIdx.x;                 // 64 channels
  float kx = kpts[2 * kp], ky = kpts[2 * kp + 1];
  float ix = kx * (160.f / 1279.f) - 0.5f;
  float iy = ky * (120.f / 959.f) - 0.5f;
  float fx0 = floorf(ix), fy0 = floorf(iy);
  int x0 = (int)fx0, y0 = (int)fy0;
  float fx = ix - fx0, fy = iy - fy0;
  const float* f = m1 + (size_t)c * 19200;
  float acc = 0.f;
  for (int dy = -1; dy < 3; ++dy) {
    float wy = cubw(fy - (float)dy);
    int yy = y0 + dy;
    for (int dx = -1; dx < 3; ++dx) {
      float wx = cubw(fx - (float)dx);
      int xx = x0 + dx;
      float v = ((unsigned)yy < 120u && (unsigned)xx < 160u) ? f[yy * 160 + xx] : 0.f;
      acc += wy * wx * v;
    }
  }
  __shared__ float sh[64];
  sh[c] = acc * acc;
  __syncthreads();
  for (int s = 32; s > 0; s >>= 1) {
    if (c < s) sh[c] += sh[c + s];
    __syncthreads();
  }
  float inv = 1.f / fmaxf(sqrtf(sh[0]), 1e-12f);
  descs[(size_t)kp * 64 + c] = acc * inv;
}

// =====================================================================
// host orchestration
// =====================================================================
extern "C" void kernel_launch(void* const* d_in, const int* in_sizes, int n_in,
                              void* d_out, int out_size, void* d_ws, size_t ws_size,
                              hipStream_t stream) {
  (void)in_sizes; (void)n_in; (void)out_size;
  const float* X = (const float*)d_in[0];
  auto P = [&](int i) { return (const float*)d_in[i]; };

  // ---- workspace carve-up (bump allocator, 256B aligned) ----
  size_t cur = 0;
  auto allocB = [&](size_t nbytes) -> void* {
    void* p = (char*)d_ws + cur;
    cur += (nbytes + 255) & ~(size_t)255;
    return p;
  };
  auto alloc = [&](size_t nfl) -> float* { return (float*)allocB(nfl * sizeof(float)); };

  float* XN   = alloc(1228800);
  float* T1   = alloc(4915200);
  float* T2   = alloc(2457600);
  float* T3   = alloc(2457600);
  float* X1   = alloc(1843200);
  float* XP   = alloc(76800);
  float* SKIP = alloc(1843200);
  float* X1S  = alloc(1843200);
  float* X2A  = alloc(1843200);
  float* X2B  = alloc(1843200);
  float* X3A  = alloc(1228800);
  float* X3B  = alloc(1228800);
  float* X3C  = alloc(1228800);
  float* X4A  = alloc(307200);
  float* X4B  = alloc(307200);
  float* X4C  = alloc(307200);
  float* X5A  = alloc(153600);
  float* X5B  = alloc(153600);
  float* X5C  = alloc(153600);
  float* X5D  = alloc(76800);
  float* M1   = alloc(1228800);
  float* HMAP = alloc(19200);
  float* PART = alloc(2048);
  float* STATS = alloc(8);

  static const int bnBase[23] = {1,6,11,16, 21,26, 31,36,41, 46,51,56, 61,66,71,76, 81,86, 93,98, 105,110,115};
  static const int bnCout[23] = {4,8,8,24, 24,24, 64,64,64, 64,64,64, 128,128,128,64, 64,64, 64,64, 64,64,64};
  float* scp[23]; float* shp[23];
  for (int i = 0; i < 23; ++i) { scp[i] = alloc(bnCout[i]); shp[i] = alloc(bnCout[i]); }
  static const int bBase[4] = {91, 103, 120, 122};   // fusionCv, heatCv, kptCv, skip1
  static const int bCout[4] = {64, 1, 65, 24};
  float* scb[4]; float* shb[4];
  for (int i = 0; i < 4; ++i) { scb[i] = alloc(bCout[i]); shb[i] = alloc(bCout[i]); }

  // ---- packed f16 weight fragments for all 27 conv layers ----
  // id 0..22 = BN layers in order; 23=fusionCv 24=heatCv 25=kptCv 26=skip1
  static const int wIdx[27]  = {5,10,15,20, 25,30, 35,40,45, 50,55,60, 65,70,75,80,
                                85,90, 97,102, 109,114,119, 92, 104, 121, 123};
  static const int wCout[27] = {4,8,8,24, 24,24, 64,64,64, 64,64,64, 128,128,128,64,
                                64,64, 64,64, 64,64,64, 64, 1, 65, 24};
  static const int wKtot[27] = {9,36,72,72, 216,216, 216,576,64, 576,576,576, 576,1152,1152,128,
                                576,576, 64,64, 64,64,64, 64, 64, 64, 1};
  _Float16* WP[27];
  for (int l = 0; l < 27; ++l) {
    size_t units = (size_t)((wCout[l] + 15) / 16) * ((wKtot[l] + 31) / 32) * 32 * 16;
    WP[l] = (_Float16*)allocB(units * sizeof(_Float16));
  }

  unsigned long long* SEL  = (unsigned long long*)alloc(8);
  unsigned*           HIST = (unsigned*)alloc(256);
  unsigned*           CNT  = (unsigned*)alloc(8);
  unsigned long long* CAND = (unsigned long long*)alloc(4096);

  if (cur > ws_size) return;   // insufficient scratch — bail deterministically

  // region reuse (dead buffers re-hosted)
  float* X4U  = T1;
  float* X5U  = T1 + 1228800;
  float* FSUM = T1 + 2457600;
  float* F1   = T1 + 3686400;
  float* F2   = T2;
  float* FEATS= T2 + 1228800;
  float* H1   = T3;
  float* H2   = T3 + 1228800;
  float* UNF  = SKIP;
  float* K1   = X1;
  float* K2   = X2A;
  float* K3   = X2B;
  float* KLOG = X1S;
  float* K1H  = X3A;
  float* LM   = X3B;
  float* REL  = X3C;
  float* SCR  = XN;

  auto conv = [&](const float* in, float* out, int lid, const float* sc, const float* sh,
                  int cin, int cout, int hin, int win, int ks, int stride, int pad, int relu) {
    int hout = (hin + 2 * pad - ks) / stride + 1;
    int wout = (win + 2 * pad - ks) / stride + 1;
    int npix = hout * wout;
    dim3 g((npix + 31) / 32, (cout + 15) / 16, 1);
    k_conv_wmma<<<g, dim3(32, 1, 1), 0, stream>>>(in, WP[lid], sc, sh, out,
        cin, cout, hin, win, hout, wout, ks, stride, pad, relu);
  };
  auto conv1 = [&](const float* in, float* out, int lid, const float* sc, const float* sh,
                   int cin, int cout, int npix, int relu) {
    dim3 g((npix + 63) / 64, (cout + 15) / 16, 1);
    k_conv1x1_async<<<g, dim3(32, 1, 1), 0, stream>>>(in, WP[lid], sc, sh, out,
        cin, cout, npix, relu);
  };

  // ---- BN folding + weight repack ----
  for (int i = 0; i < 23; ++i)
    k_bn_prep<<<1, 128, 0, stream>>>(P(bnBase[i] + 1), P(bnBase[i]), P(bnBase[i] + 2),
                                     P(bnBase[i] + 3), scp[i], shp[i], bnCout[i]);
  for (int i = 0; i < 4; ++i)
    k_bias_prep<<<1, 128, 0, stream>>>(P(bBase[i]), scb[i], shb[i], bCout[i]);
  for (int l = 0; l < 27; ++l) {
    int total = ((wCout[l] + 15) / 16) * ((wKtot[l] + 31) / 32) * 32;
    k_repack<<<(total + 127) / 128, 128, 0, stream>>>(P(wIdx[l]), WP[l], wCout[l], wKtot[l]);
  }

  // ---- instance norm ----
  k_stats_partial<<<1024, 256, 0, stream>>>(X, 1228800, PART);
  k_stats_final<<<1, 256, 0, stream>>>(PART, STATS, 1228800);
  k_instnorm<<<4800, 256, 0, stream>>>(X, STATS, XN, 1228800);

  // ---- backbone ----
  conv(XN, T1, 0, scp[0], shp[0], 1, 4, 960, 1280, 3, 1, 1, 1);
  conv(T1, T2, 1, scp[1], shp[1], 4, 8, 960, 1280, 3, 2, 1, 1);
  conv(T2, T3, 2, scp[2], shp[2], 8, 8, 480, 640, 3, 1, 1, 1);
  conv(T3, X1, 3, scp[3], shp[3], 8, 24, 480, 640, 3, 2, 1, 1);

  k_avgpool4<<<(76800 + 255) / 256, 256, 0, stream>>>(XN, XP, 960, 1280);
  conv(XP, SKIP, 26, scb[3], shb[3], 1, 24, 240, 320, 1, 1, 0, 0);
  k_add<<<(1843200 + 255) / 256, 256, 0, stream>>>(X1, SKIP, X1S, 1843200);

  conv(X1S, X2A, 4, scp[4], shp[4], 24, 24, 240, 320, 3, 1, 1, 1);
  conv(X2A, X2B, 5, scp[5], shp[5], 24, 24, 240, 320, 3, 1, 1, 1);

  conv(X2B, X3A, 6, scp[6], shp[6], 24, 64, 240, 320, 3, 2, 1, 1);
  conv(X3A, X3B, 7, scp[7], shp[7], 64, 64, 120, 160, 3, 1, 1, 1);
  conv1(X3B, X3C, 8, scp[8], shp[8], 64, 64, 19200, 1);

  conv(X3C, X4A, 9,  scp[9],  shp[9],  64, 64, 120, 160, 3, 2, 1, 1);
  conv(X4A, X4B, 10, scp[10], shp[10], 64, 64, 60, 80, 3, 1, 1, 1);
  conv(X4B, X4C, 11, scp[11], shp[11], 64, 64, 60, 80, 3, 1, 1, 1);

  conv(X4C, X5A, 12, scp[12], shp[12], 64, 128, 60, 80, 3, 2, 1, 1);
  conv(X5A, X5B, 13, scp[13], shp[13], 128, 128, 30, 40, 3, 1, 1, 1);
  conv(X5B, X5C, 14, scp[14], shp[14], 128, 128, 30, 40, 3, 1, 1, 1);
  conv1(X5C, X5D, 15, scp[15], shp[15], 128, 64, 1200, 1);

  k_resize<<<4800, 256, 0, stream>>>(X4C, X4U, 64, 60, 80, 120, 160);
  k_resize<<<4800, 256, 0, stream>>>(X5D, X5U, 64, 30, 40, 120, 160);
  k_add3<<<4800, 256, 0, stream>>>(X3C, X4U, X5U, FSUM, 1228800);

  conv(FSUM, F1, 16, scp[16], shp[16], 64, 64, 120, 160, 3, 1, 1, 1);
  conv(F1, F2, 17, scp[17], shp[17], 64, 64, 120, 160, 3, 1, 1, 1);
  conv1(F2, FEATS, 23, scb[0], shb[0], 64, 64, 19200, 0);

  conv1(FEATS, H1, 18, scp[18], shp[18], 64, 64, 19200, 1);
  conv1(H1, H2, 19, scp[19], shp[19], 64, 64, 19200, 1);
  conv1(H2, HMAP, 24, scb[1], shb[1], 64, 1, 19200, 0);
  k_sigmoid<<<(19200 + 255) / 256, 256, 0, stream>>>(HMAP, 19200);

  k_unfold8<<<4800, 256, 0, stream>>>(XN, UNF);
  conv1(UNF, K1, 20, scp[20], shp[20], 64, 64, 19200, 1);
  conv1(K1, K2, 21, scp[21], shp[21], 64, 64, 19200, 1);
  conv1(K2, K3, 22, scp[22], shp[22], 64, 64, 19200, 1);
  conv1(K3, KLOG, 25, scb[2], shb[2], 64, 65, 19200, 0);

  // ---- detection post-processing ----
  k_kpts_heat<<<(19200 + 255) / 256, 256, 0, stream>>>(KLOG, K1H);
  k_maxpool5<<<4800, 256, 0, stream>>>(K1H, LM, 960, 1280);
  k_rel<<<4800, 256, 0, stream>>>(HMAP, REL);
  k_scores<<<4800, 256, 0, stream>>>(K1H, LM, REL, SCR, 1228800);
  k_normfeat<<<(19200 + 255) / 256, 256, 0, stream>>>(FEATS, M1);

  // ---- deterministic top-2000 ----
  k_sel_init<<<1, 1, 0, stream>>>(SEL);
  for (int lvl = 0; lvl < 8; ++lvl) {
    hipMemsetAsync(HIST, 0, 256 * sizeof(unsigned), stream);
    k_hist64<<<1200, 256, 0, stream>>>(SCR, 1228800, SEL, lvl, HIST);
    k_pick64<<<1, 1, 0, stream>>>(HIST, SEL);
  }
  hipMemsetAsync(CAND, 0, 2048 * sizeof(unsigned long long), stream);
  hipMemsetAsync(CNT, 0, sizeof(unsigned), stream);
  k_compact64<<<1200, 256, 0, stream>>>(SCR, 1228800, SEL, CAND, CNT);
  k_bitonic2048<<<1, 1024, 0, stream>>>(CAND);

  float* OMK = (float*)d_out;        // (1,2000,2)
  float* OSC = OMK + 4000;           // (1,2000)
  float* ODS = OMK + 6000;           // (1,2000,64)
  k_emit<<<(2000 + 255) / 256, 256, 0, stream>>>(CAND, OMK, OSC);
  k_desc<<<2000, 64, 0, stream>>>(M1, OMK, ODS);
}
